// SSTV3_81819126989497
// MI455X (gfx1250) — compile-verified, matrix operand order checked by reading, not run on hardware
//
#include <hip/hip_runtime.h>
#include <hip/hip_bf16.h>
#include <math.h>

typedef _Float16 half_t;
typedef __attribute__((ext_vector_type(16))) _Float16 v16h;
typedef __attribute__((ext_vector_type(8)))  float    v8f;

#define NB   128   // B  (videos)
#define NBC  128   // BC (captions)
#define NTV  20
#define NTC  20
#define NFC  2048
#define NVEMB 512
#define NWEMB 512
#define NH   512
#define NATT 256

#define LDSP 40    // padded LDS K-stride in halves (32 data + 8 pad)

// ---------------------------------------------------------------------------
// WMMA fragment load from an LDS panel laid out [rows][LDSP] (K-major, f16).
// A-matrix 16x32 f16 layout (CDNA5 ISA 7.12.2):
//   lane L<16 : row M=L,   holds K = 0..7  and 16..23
//   lane L>=16: row M=L-16, holds K = 8..15 and 24..31
// B fragment loaded from a B^T panel with the identical layout (N plays M).
// ---------------------------------------------------------------------------
__device__ __forceinline__ v16h frag_ld(const half_t* p, int rowbase, int lane) {
    v16h f;
    int row  = rowbase + (lane & 15);
    int koff = (lane >> 4) << 3;            // 0 or 8
    const half_t* q = p + row * LDSP;
#pragma unroll
    for (int e = 0; e < 16; ++e) {
        int k = ((e < 8) ? e : e + 8) + koff;
        f[e] = q[k];
    }
    return f;
}

__device__ __forceinline__ v8f v8f_zero() {
    v8f z = {0.f,0.f,0.f,0.f,0.f,0.f,0.f,0.f};
    return z;
}

__device__ __forceinline__ float sigmoidf_(float x) {
    return 1.f / (1.f + __expf(-x));
}

// LDS offset of a __shared__ object: AS3->generic cast keeps the workgroup-
// relative LDS byte offset in the low 32 bits of the flat address.
__device__ __forceinline__ unsigned lds_off_of(const void* p) {
    return (unsigned)(uintptr_t)p;
}

// CDNA5 async copy: per-lane 16B global -> LDS, tracked by ASYNCcnt.
__device__ __forceinline__ void cp_async_b128(unsigned lds_addr, const void* gaddr) {
    asm volatile("global_load_async_to_lds_b128 %0, %1, off"
                 :: "v"(lds_addr), "v"((unsigned long long)(uintptr_t)gaddr)
                 : "memory");
}
__device__ __forceinline__ void wait_async0() {
    asm volatile("s_wait_asynccnt 0" ::: "memory");
}

// ---------------------------------------------------------------------------
// f32 -> f16 convert
// ---------------------------------------------------------------------------
__global__ void k_cvt_h(const float* __restrict__ s, half_t* __restrict__ d, int n) {
    int i = blockIdx.x * 256 + threadIdx.x;
    if (i < n) d[i] = (half_t)s[i];
}

// ---------------------------------------------------------------------------
// Stage a 64x32-half tile into an LDS panel (LDSP layout) with async b128
// copies: 256 transfers, 128 threads -> 2 per thread. Rows must be 16B
// aligned and in range (vector path is only used when M,N%64==0, ld%8==0).
// ---------------------------------------------------------------------------
__device__ __forceinline__ void stage64_async(unsigned lds_base, const half_t* src,
                                              int ld, int row0, int kbase, int tid) {
#pragma unroll
    for (int v = 0; v < 2; ++v) {
        int idx = tid + (v << 7);
        int r  = idx >> 2;
        int cq = (idx & 3) << 3;
        cp_async_b128(lds_base + (unsigned)(r * LDSP + cq) * 2,
                      src + (size_t)(row0 + r) * ld + kbase + cq);
    }
}

// ---------------------------------------------------------------------------
// Generic WMMA GEMM: C[M,N] = A[M,K](f16) * B^T[N,K](f16) + bias1 + bias2
// Block tile 64x64, 4 waves in 2x2, each wave 32x32 = 2x2 WMMA C tiles.
// Fast path (K%32==0, aligned rows): double-buffered async-to-LDS staging.
// Slow path (ragged K, e.g. 300): scalar bounds-checked staging.
// ---------------------------------------------------------------------------
__global__ __launch_bounds__(128)
void k_gemm_f16(const half_t* __restrict__ A, const half_t* __restrict__ Bt,
                const float* __restrict__ bias1, const float* __restrict__ bias2,
                float* __restrict__ Cf, half_t* __restrict__ Ch,
                int M, int N, int K, int lda, int ldb)
{
    __shared__ half_t Asb[2][64 * LDSP];
    __shared__ half_t Bsb[2][64 * LDSP];

    int tid  = threadIdx.x;
    int lane = tid & 31;
    int wave = tid >> 5;
    int wm   = (wave >> 1) * 32;
    int wn   = (wave & 1) * 32;
    int mbase = blockIdx.y * 64;
    int nbase = blockIdx.x * 64;

    v8f acc[2][2];
    acc[0][0] = v8f_zero(); acc[0][1] = v8f_zero();
    acc[1][0] = v8f_zero(); acc[1][1] = v8f_zero();

    int nk = (K + 31) >> 5;
    bool vec = ((K & 31) == 0) && ((lda & 7) == 0) && ((ldb & 7) == 0);

    if (vec) {
        unsigned a0 = lds_off_of(Asb[0]), a1 = lds_off_of(Asb[1]);
        unsigned b0 = lds_off_of(Bsb[0]), b1 = lds_off_of(Bsb[1]);
        stage64_async(a0, A,  lda, mbase, 0, tid);
        stage64_async(b0, Bt, ldb, nbase, 0, tid);
        int cur = 0;
        for (int kc = 0; kc < nk; ++kc) {
            wait_async0();                 // this wave's copies into buf[cur] done
            __syncthreads();               // all waves' copies done; prior reads done
            if (kc + 1 < nk) {             // prefetch next K-chunk into other buffer
                stage64_async(cur ? a0 : a1, A,  lda, mbase, (kc + 1) << 5, tid);
                stage64_async(cur ? b0 : b1, Bt, ldb, nbase, (kc + 1) << 5, tid);
            }
            const half_t* As = Asb[cur];
            const half_t* Bs = Bsb[cur];
            v16h af0 = frag_ld(As, wm,      lane);
            v16h af1 = frag_ld(As, wm + 16, lane);
            v16h bf0 = frag_ld(Bs, wn,      lane);
            v16h bf1 = frag_ld(Bs, wn + 16, lane);
            acc[0][0] = __builtin_amdgcn_wmma_f32_16x16x32_f16(false, af0, false, bf0, (short)0, acc[0][0], false, false);
            acc[0][1] = __builtin_amdgcn_wmma_f32_16x16x32_f16(false, af0, false, bf1, (short)0, acc[0][1], false, false);
            acc[1][0] = __builtin_amdgcn_wmma_f32_16x16x32_f16(false, af1, false, bf0, (short)0, acc[1][0], false, false);
            acc[1][1] = __builtin_amdgcn_wmma_f32_16x16x32_f16(false, af1, false, bf1, (short)0, acc[1][1], false, false);
            cur ^= 1;
        }
    } else {
        for (int kc = 0; kc < nk; ++kc) {
            int kbase = kc << 5;
            __syncthreads();
            for (int i = tid; i < 64 * 32; i += 128) {
                int r = i >> 5, c = i & 31;
                int gk = kbase + c;
                half_t av = (half_t)0.f, bv = (half_t)0.f;
                if (gk < K) {
                    int gm = mbase + r; if (gm < M) av = A [(size_t)gm * lda + gk];
                    int gn = nbase + r; if (gn < N) bv = Bt[(size_t)gn * ldb + gk];
                }
                Asb[0][r * LDSP + c] = av;
                Bsb[0][r * LDSP + c] = bv;
            }
            __syncthreads();
            v16h af0 = frag_ld(Asb[0], wm,      lane);
            v16h af1 = frag_ld(Asb[0], wm + 16, lane);
            v16h bf0 = frag_ld(Bsb[0], wn,      lane);
            v16h bf1 = frag_ld(Bsb[0], wn + 16, lane);
            acc[0][0] = __builtin_amdgcn_wmma_f32_16x16x32_f16(false, af0, false, bf0, (short)0, acc[0][0], false, false);
            acc[0][1] = __builtin_amdgcn_wmma_f32_16x16x32_f16(false, af0, false, bf1, (short)0, acc[0][1], false, false);
            acc[1][0] = __builtin_amdgcn_wmma_f32_16x16x32_f16(false, af1, false, bf0, (short)0, acc[1][0], false, false);
            acc[1][1] = __builtin_amdgcn_wmma_f32_16x16x32_f16(false, af1, false, bf1, (short)0, acc[1][1], false, false);
        }
    }

#pragma unroll
    for (int i = 0; i < 2; ++i) {
#pragma unroll
        for (int j = 0; j < 2; ++j) {
            int row0 = mbase + wm + i * 16 + ((lane >> 4) << 3);
            int col  = nbase + wn + j * 16 + (lane & 15);
            float bsum = 0.f;
            if (bias1) bsum += bias1[col];
            if (bias2) bsum += bias2[col];
#pragma unroll
            for (int r = 0; r < 8; ++r) {
                int m = row0 + r;
                float v = acc[i][j][r] + bsum;
                size_t idx = (size_t)m * N + col;
                if (Cf) Cf[idx] = v;
                if (Ch) Ch[idx] = (half_t)v;
            }
        }
    }
}

// ---------------------------------------------------------------------------
// One LSTM time step. gates_rec = h_prev @ W_hh^T via WMMA with all 4 gate
// accumulators in registers (acc[4][2]); cell update reads gates straight
// from the accumulators. Double-buffered async staging of h/W_hh panels.
// Block tile: 32 rows (BC) x 64 cols (H); grid = (NH/64, NBC/32).
// ---------------------------------------------------------------------------
__global__ __launch_bounds__(128)
void k_lstm_step(const half_t* __restrict__ h_prev, half_t* __restrict__ h_next,
                 float* __restrict__ c, const float* __restrict__ gates_x,
                 const half_t* __restrict__ Whh,
                 float* __restrict__ capouts, half_t* __restrict__ capouts_h,
                 int t)
{
    __shared__ half_t Asb[2][32 * LDSP];
    __shared__ half_t Bsb[2][4 * 64 * LDSP];

    int tid  = threadIdx.x;
    int lane = tid & 31;
    int w    = tid >> 5;               // wave 0..3 -> 16-col strip
    int nb   = blockIdx.x * 64;        // H col base
    int mbase = blockIdx.y * 32;       // BC row base

    v8f acc[4][2];
#pragma unroll
    for (int g = 0; g < 4; ++g) { acc[g][0] = v8f_zero(); acc[g][1] = v8f_zero(); }

    unsigned a0 = lds_off_of(Asb[0]), a1 = lds_off_of(Asb[1]);
    unsigned b0 = lds_off_of(Bsb[0]), b1 = lds_off_of(Bsb[1]);

    // stage(kbase): A = 32x32 (1 copy/thread), B = 4 gates x 64x32 (8/thread)
    auto stage = [&](unsigned asb, unsigned bsb, int kbase) {
        {
            int r = tid >> 2, cq = (tid & 3) << 3;
            cp_async_b128(asb + (unsigned)(r * LDSP + cq) * 2,
                          h_prev + (size_t)(mbase + r) * NH + kbase + cq);
        }
#pragma unroll
        for (int v = 0; v < 8; ++v) {
            int idx = tid + (v << 7);
            int g  = idx >> 8;
            int rr = (idx & 255) >> 2;
            int cq = (idx & 3) << 3;
            cp_async_b128(bsb + (unsigned)((g * 64 + rr) * LDSP + cq) * 2,
                          Whh + (size_t)(g * NH + nb + rr) * NH + kbase + cq);
        }
    };

    stage(a0, b0, 0);
    int cur = 0;
    for (int kc = 0; kc < NH / 32; ++kc) {
        wait_async0();
        __syncthreads();
        if (kc + 1 < NH / 32)
            stage(cur ? a0 : a1, cur ? b0 : b1, (kc + 1) << 5);
        const half_t* As = Asb[cur];
        const half_t* Bs = Bsb[cur];
        v16h af0 = frag_ld(As, 0,  lane);
        v16h af1 = frag_ld(As, 16, lane);
#pragma unroll
        for (int g = 0; g < 4; ++g) {
            v16h bf = frag_ld(Bs, g * 64 + w * 16, lane);
            acc[g][0] = __builtin_amdgcn_wmma_f32_16x16x32_f16(false, af0, false, bf, (short)0, acc[g][0], false, false);
            acc[g][1] = __builtin_amdgcn_wmma_f32_16x16x32_f16(false, af1, false, bf, (short)0, acc[g][1], false, false);
        }
        cur ^= 1;
    }

    // pointwise LSTM cell update straight from accumulator registers
    int n  = nb + w * 16 + (lane & 15);          // hidden index
    int rh = (lane >> 4) << 3;
#pragma unroll
    for (int i = 0; i < 2; ++i) {
#pragma unroll
        for (int r = 0; r < 8; ++r) {
            int m = mbase + i * 16 + rh + r;     // caption index j
            const float* gx = gates_x + (size_t)(m * NTC + t) * (4 * NH);
            float ig = sigmoidf_(acc[0][i][r] + gx[n]);
            float fg = sigmoidf_(acc[1][i][r] + gx[n + NH]);
            float gg = tanhf   (acc[2][i][r] + gx[n + 2 * NH]);
            float og = sigmoidf_(acc[3][i][r] + gx[n + 3 * NH]);
            float cold = c[m * NH + n];
            float cn = fg * cold + ig * gg;
            float hn = og * tanhf(cn);
            c[m * NH + n]      = cn;
            h_next[m * NH + n] = (half_t)hn;
            size_t oi = (size_t)(m * NTC + t) * NH + n;
            capouts[oi]   = hn;
            capouts_h[oi] = (half_t)hn;
        }
    }
}

// ---------------------------------------------------------------------------
// Mean-pool video features over TV, store f16 for the vfc GEMM.
// ---------------------------------------------------------------------------
__global__ void k_vmean(const float* __restrict__ vf, half_t* __restrict__ vmean_h) {
    int i = blockIdx.x * 256 + threadIdx.x;       // over NB*NFC
    if (i < NB * NFC) {
        int b = i >> 11, k = i & (NFC - 1);
        const float* p = vf + (size_t)b * NTV * NFC + k;
        float s = 0.f;
#pragma unroll
        for (int t = 0; t < NTV; ++t) s += p[t * NFC];
        vmean_h[i] = (half_t)(s * (1.f / NTV));
    }
}

// ---------------------------------------------------------------------------
// nv[b] = max(||vfc[b]||, eps)
// ---------------------------------------------------------------------------
__global__ __launch_bounds__(256)
void k_nv(const float* __restrict__ vfc, float* __restrict__ nv) {
    __shared__ float red[256];
    int b = blockIdx.x, tid = threadIdx.x;
    float s = 0.f;
    for (int d = tid; d < NH; d += 256) { float v = vfc[b * NH + d]; s += v * v; }
    red[tid] = s; __syncthreads();
    for (int o = 128; o > 0; o >>= 1) { if (tid < o) red[tid] += red[tid + o]; __syncthreads(); }
    if (tid == 0) nv[b] = fmaxf(sqrtf(red[0]), 1e-6f);
}

// ---------------------------------------------------------------------------
// Fused attention tail per (b, j): tanh-scores, masked softmax over t,
// weighted sum over cap_outs, cosine similarity.
// ---------------------------------------------------------------------------
__global__ __launch_bounds__(256)
void k_att(const float* __restrict__ vlin, const float* __restrict__ clin,
           const float* __restrict__ capouts, const float* __restrict__ vfc,
           const float* __restrict__ nv, const float* __restrict__ W_att,
           const int* __restrict__ cap_len, float* __restrict__ out)
{
    __shared__ float vl[NATT], wa[NATT], sc[NTC], wts[NTC];
    __shared__ float redA[256], redB[256];
    int j = blockIdx.x, b = blockIdx.y;
    int tid = threadIdx.x, lane = tid & 31, wave = tid >> 5;

    vl[tid] = vlin[b * NATT + tid];
    wa[tid] = W_att[tid];
    __syncthreads();

    for (int t = wave; t < NTC; t += 8) {
        const float* cl = clin + (size_t)(j * NTC + t) * NATT;
        float s = 0.f;
        for (int q = lane; q < NATT; q += 32) s += tanhf(vl[q] + cl[q]) * wa[q];
        for (int o = 16; o > 0; o >>= 1) s += __shfl_down(s, o, 32);
        if (lane == 0) sc[t] = s;
    }
    __syncthreads();

    if (tid == 0) {
        int L = cap_len[j]; L = L < 1 ? 1 : (L > NTC ? NTC : L);
        float mx = -3.0e38f;
        for (int t = 0; t < L; ++t) mx = fmaxf(mx, sc[t]);
        float sum = 0.f;
        for (int t = 0; t < L; ++t) { float e = __expf(sc[t] - mx); wts[t] = e; sum += e; }
        float inv = 1.f / sum;
        for (int t = 0; t < NTC; ++t) wts[t] = (t < L) ? wts[t] * inv : 0.f;
    }
    __syncthreads();

    float dv = 0.f, nr = 0.f;
    for (int d = tid; d < NH; d += 256) {
        const float* co = capouts + (size_t)(j * NTC) * NH + d;
        float a = 0.f;
#pragma unroll
        for (int t = 0; t < NTC; ++t) a += wts[t] * co[t * NH];
        dv += a * vfc[b * NH + d];
        nr += a * a;
    }
    redA[tid] = dv; redB[tid] = nr; __syncthreads();
    for (int o = 128; o > 0; o >>= 1) {
        if (tid < o) { redA[tid] += redA[tid + o]; redB[tid] += redB[tid + o]; }
        __syncthreads();
    }
    if (tid == 0) {
        float na = fmaxf(sqrtf(redB[0]), 1e-6f);
        out[b * NBC + j] = redA[0] / (na * nv[b]);
    }
}

// ---------------------------------------------------------------------------

extern "C" void kernel_launch(void* const* d_in, const int* in_sizes, int n_in,
                              void* d_out, int out_size, void* d_ws, size_t ws_size,
                              hipStream_t stream) {
    const float* video_fc = (const float*)d_in[0];
    const float* caption  = (const float*)d_in[1];
    const int*   cap_len  = (const int*)  d_in[2];
    const float* W_word = (const float*)d_in[3];  const float* b_word = (const float*)d_in[4];
    const float* W_ih   = (const float*)d_in[5];  const float* b_ih   = (const float*)d_in[6];
    const float* W_hh   = (const float*)d_in[7];  const float* b_hh   = (const float*)d_in[8];
    const float* W_vemb = (const float*)d_in[9];  const float* b_vemb = (const float*)d_in[10];
    const float* W_vlin = (const float*)d_in[11]; const float* b_vlin = (const float*)d_in[12];
    const float* W_slin = (const float*)d_in[13]; const float* b_slin = (const float*)d_in[14];
    const float* W_att  = (const float*)d_in[15]; // b_att: softmax-invariant, skipped
    float* out = (float*)d_out;

    char*  base = (char*)d_ws;
    size_t off  = 0;
    auto take = [&](size_t bytes) -> char* {
        char* p = base + off;
        off += (bytes + 255) & ~(size_t)255;
        return p;
    };

    half_t* Wword_h  = (half_t*)take((size_t)NWEMB * 300 * 2);
    half_t* Wih_h    = (half_t*)take((size_t)4 * NH * NH * 2);
    half_t* Whh_h    = (half_t*)take((size_t)4 * NH * NH * 2);
    half_t* Wvemb_h  = (half_t*)take((size_t)NVEMB * NFC * 2);
    half_t* Wvlin_h  = (half_t*)take((size_t)NATT * NVEMB * 2);
    half_t* Wslin_h  = (half_t*)take((size_t)NATT * NH * 2);
    half_t* cap_h    = (half_t*)take((size_t)NBC * NTC * 300 * 2);
    half_t* xemb_h   = (half_t*)take((size_t)NBC * NTC * NWEMB * 2);
    float*  gates_x  = (float*) take((size_t)NBC * NTC * 4 * NH * 4);
    half_t* vmean_h  = (half_t*)take((size_t)NB * NFC * 2);
    float*  vfc      = (float*) take((size_t)NB * NVEMB * 4);
    half_t* vfc_h    = (half_t*)take((size_t)NB * NVEMB * 2);
    float*  vlin     = (float*) take((size_t)NB * NATT * 4);
    float*  capouts  = (float*) take((size_t)NBC * NTC * NH * 4);
    half_t* capout_h = (half_t*)take((size_t)NBC * NTC * NH * 2);
    float*  clin     = (float*) take((size_t)NBC * NTC * NATT * 4);
    half_t* h_a      = (half_t*)take((size_t)NBC * NH * 2);
    half_t* h_b      = (half_t*)take((size_t)NBC * NH * 2);
    float*  c_st     = (float*) take((size_t)NBC * NH * 4);
    float*  nv       = (float*) take((size_t)NB * 4);
    (void)ws_size; (void)in_sizes; (void)n_in; (void)out_size;

    auto cvt = [&](const float* s, half_t* d, int n) {
        k_cvt_h<<<dim3((n + 255) / 256), dim3(256), 0, stream>>>(s, d, n);
    };
    // 1. weight + input conversions to f16
    cvt(W_word, Wword_h, NWEMB * 300);
    cvt(W_ih,   Wih_h,   4 * NH * NH);
    cvt(W_hh,   Whh_h,   4 * NH * NH);
    cvt(W_vemb, Wvemb_h, NVEMB * NFC);
    cvt(W_vlin, Wvlin_h, NATT * NVEMB);
    cvt(W_slin, Wslin_h, NATT * NH);
    cvt(caption, cap_h,  NBC * NTC * 300);

    // 2. word embedding: xemb[2560,512] = caption @ W_word^T + b_word (f16 out)
    k_gemm_f16<<<dim3(NWEMB / 64, (NBC * NTC) / 64), dim3(128), 0, stream>>>(
        cap_h, Wword_h, b_word, nullptr, nullptr, xemb_h,
        NBC * NTC, NWEMB, 300, 300, 300);

    // 3. x-path gates for ALL timesteps: gates_x = xemb @ W_ih^T + b_ih + b_hh
    k_gemm_f16<<<dim3((4 * NH) / 64, (NBC * NTC) / 64), dim3(128), 0, stream>>>(
        xemb_h, Wih_h, b_ih, b_hh, gates_x, nullptr,
        NBC * NTC, 4 * NH, NWEMB, NWEMB, NWEMB);

    // 4. mean-pool video features; 5. vfc = vmean @ W_vemb^T + b_vemb
    k_vmean<<<dim3((NB * NFC + 255) / 256), dim3(256), 0, stream>>>(video_fc, vmean_h);
    k_gemm_f16<<<dim3(NVEMB / 64, NB / 64), dim3(128), 0, stream>>>(
        vmean_h, Wvemb_h, b_vemb, nullptr, vfc, vfc_h,
        NB, NVEMB, NFC, NFC, NFC);

    // 6. vlin = vfc @ W_vlin^T + b_vlin
    k_gemm_f16<<<dim3(NATT / 64, NB / 64), dim3(128), 0, stream>>>(
        vfc_h, Wvlin_h, b_vlin, nullptr, vlin, nullptr,
        NB, NATT, NVEMB, NVEMB, NVEMB);

    // 7. LSTM: zero h,c then 20 sequential steps (h double-buffered)
    hipMemsetAsync(h_a,  0, (size_t)NBC * NH * 2, stream);
    hipMemsetAsync(c_st, 0, (size_t)NBC * NH * 4, stream);
    for (int t = 0; t < NTC; ++t) {
        half_t* hp = (t & 1) ? h_b : h_a;
        half_t* hn = (t & 1) ? h_a : h_b;
        k_lstm_step<<<dim3(NH / 64, NBC / 32), dim3(128), 0, stream>>>(
            hp, hn, c_st, gates_x, Whh_h, capouts, capout_h, t);
    }

    // 8. clin = cap_outs @ W_slin^T + b_slin
    k_gemm_f16<<<dim3(NATT / 64, (NBC * NTC) / 64), dim3(128), 0, stream>>>(
        capout_h, Wslin_h, b_slin, nullptr, clin, nullptr,
        NBC * NTC, NATT, NH, NH, NH);

    // 9. norms of vfc; 10. fused attention + softmax + cosine similarity
    k_nv<<<dim3(NB), dim3(256), 0, stream>>>(vfc, nv);
    k_att<<<dim3(NBC, NB), dim3(256), 0, stream>>>(
        vlin, clin, capouts, vfc, nv, W_att, cap_len, out);
}
